// GAT_22969485099999
// MI455X (gfx1250) — matrix operand order repaired and checked by reading, not compiled, not run on hardware
//
#include <hip/hip_runtime.h>
#include <hip/hip_bf16.h>

// ---------------- problem constants (from reference) ----------------
#define NN   50000      // nodes
#define EE   800000     // edges before self-loops
#define ETOT (EE + NN)  // edges incl self-loops
#define KDIM 128        // K of every projection (IN = HEADS*HID = 128)

typedef __attribute__((ext_vector_type(16))) _Float16 v16h;
typedef __attribute__((ext_vector_type(8)))  _Float16 v8h;
typedef __attribute__((ext_vector_type(8)))  float    v8f;

// ---------------- helpers ----------------
__device__ __forceinline__ unsigned fenc(float f) {
    unsigned u = __float_as_uint(f);
    return (u & 0x80000000u) ? ~u : (u | 0x80000000u);   // monotonic float->uint
}
__device__ __forceinline__ float fdec(unsigned u) {
    return (u & 0x80000000u) ? __uint_as_float(u ^ 0x80000000u) : __uint_as_float(~u);
}

// ---------------- f32 -> f16 row-major copy (A operand, once per layer) ----------------
__global__ void gat_cvt16(const float* __restrict__ src, _Float16* __restrict__ dst, int n4)
{
    int t = blockIdx.x * blockDim.x + threadIdx.x;
    if (t >= n4) return;
    float4 v = ((const float4*)src)[t];
    v8h* d2 = nullptr; (void)d2;
    _Float16* d = dst + (size_t)t * 4;
    d[0] = (_Float16)v.x; d[1] = (_Float16)v.y; d[2] = (_Float16)v.z; d[3] = (_Float16)v.w;
}

// ---------------- pack W[128,Nc] into WMMA B-fragments (f16), once per layer ----------
// fragment index: ((kb32 * ntn + tn) * 32 + lane) * 16 halfs
// B 32x16 f16 lane order used here (same k order as the A loads): N = lane&15, K = kb + (lane>>4)*16 + j
__global__ void gat_packB(const float* __restrict__ B, _Float16* __restrict__ P, int Nc)
{
    int ntn = Nc >> 4;
    int total = (KDIM / 32) * ntn * 32;
    int t = blockIdx.x * blockDim.x + threadIdx.x;
    if (t >= total) return;
    int lane = t & 31;
    int tile = t >> 5;                 // kb32 * ntn + tn
    int tn   = tile % ntn;
    int kb   = (tile / ntn) * 32;
    int half = lane >> 4, l15 = lane & 15;
    _Float16* dst = P + (size_t)t * 16;
    const float* bcol = B + (size_t)(kb + half * 16) * Nc + tn * 16 + l15;
    #pragma unroll
    for (int j = 0; j < 16; ++j) dst[j] = (_Float16)bcol[j * Nc];
}

// ---------------- WMMA GEMM: C[M,Nc] = A16[M,128] * Bpack ----------------
// one wave (32 threads) per 16x16 output tile; K loop fully unrolled -> 4 WMMAs
__global__ __launch_bounds__(32) void gat_gemm_wmma(
    const _Float16* __restrict__ A16, const _Float16* __restrict__ Bpack,
    float* __restrict__ C, int Nc)
{
    const int ntn  = Nc >> 4;
    const int tm   = blockIdx.x / ntn;
    const int tn   = blockIdx.x % ntn;
    const int lane = threadIdx.x;
    const int half = lane >> 4;       // 0 | 1
    const int l15  = lane & 15;

    const int m0 = tm << 4;
    const int n0 = tn << 4;

    v8f acc = {};
    const _Float16* arow = A16 + (size_t)(m0 + l15) * KDIM + half * 16;

    #pragma unroll
    for (int kb = 0; kb < KDIM; kb += 32) {
        // A 16x32 f16 layout: two contiguous 8-half chunks per lane (16B aligned each)
        v8h lo = *(const v8h*)(arow + kb);
        v8h hi = *(const v8h*)(arow + kb + 8);
        v16h a;
        #pragma unroll
        for (int j = 0; j < 8; ++j) { a[j] = lo[j]; a[8 + j] = hi[j]; }

        // B fragment: one aligned 32B vector load from the packed buffer
        v16h b = *(const v16h*)(Bpack +
                  ((size_t)(((kb >> 5) * ntn + tn) * 32 + lane)) * 16);

        acc = __builtin_amdgcn_wmma_f32_16x16x32_f16(
                  false, a, false, b, (short)0, acc, false, false);
    }

    // C/D layout: VGPR v -> M = v + half*8, N = l15
    #pragma unroll
    for (int v = 0; v < 8; ++v)
        C[(size_t)(m0 + v + half * 8) * Nc + n0 + l15] = acc[v];
}

// ---------------- attention logit dots: al[n,h] = sum_c g[n,h,c]*a[h,c] ----------------
__global__ void gat_al(const float* __restrict__ g,
                       const float* __restrict__ a_src, const float* __restrict__ a_dst,
                       float* __restrict__ al_s, float* __restrict__ al_d,
                       int heads, int Cout)
{
    int t = blockIdx.x * blockDim.x + threadIdx.x;
    if (t >= NN * heads) return;
    int node = t / heads, h = t % heads;
    const float* row = g + (size_t)(node * heads + h) * Cout;
    const float* as  = a_src + h * Cout;
    const float* ad  = a_dst + h * Cout;
    float ss = 0.f, sd = 0.f;
    for (int c = 0; c < Cout; ++c) { float v = row[c]; ss += v * as[c]; sd += v * ad[c]; }
    al_s[t] = ss; al_d[t] = sd;
}

// ---------------- zero init ----------------
__global__ void gat_zero(unsigned* __restrict__ p, int n)
{
    int t = blockIdx.x * blockDim.x + threadIdx.x;
    if (t < n) p[t] = 0u;
}

// ---------------- per-(edge,head): logits + leaky relu + segment max ----------------
__global__ void gat_edge_logits(const int* __restrict__ ei,
                                const float* __restrict__ al_s, const float* __restrict__ al_d,
                                float* __restrict__ e_out, unsigned* __restrict__ m_enc,
                                int heads)
{
    int t = blockIdx.x * blockDim.x + threadIdx.x;
    if (t >= ETOT * heads) return;
    int i = t / heads, h = t % heads;
    int src = (i < EE) ? ei[i]      : (i - EE);
    int dst = (i < EE) ? ei[EE + i] : (i - EE);
    float e = al_s[src * heads + h] + al_d[dst * heads + h];
    e = (e > 0.f) ? e : 0.2f * e;                 // LeakyReLU(0.2)
    e_out[t] = e;
    atomicMax(&m_enc[dst * heads + h], fenc(e));  // segment max (monotonic encoding)
}

// ---------------- per-(edge,head): exp(e - m[dst]) + segment sum ----------------
__global__ void gat_edge_exp(const int* __restrict__ ei,
                             float* __restrict__ e_buf,
                             const unsigned* __restrict__ m_enc,
                             float* __restrict__ s_sum, int heads)
{
    int t = blockIdx.x * blockDim.x + threadIdx.x;
    if (t >= ETOT * heads) return;
    int i = t / heads, h = t % heads;
    int dst = (i < EE) ? ei[EE + i] : (i - EE);
    float m  = fdec(m_enc[dst * heads + h]);
    float ee = expf(e_buf[t] - m);
    e_buf[t] = ee;
    atomicAdd(&s_sum[dst * heads + h], ee);
}

// ---------------- aggregation: acc[dst] += g[src] * alpha ; one wave32 per edge ----------------
__global__ void gat_edge_agg(const int* __restrict__ ei,
                             const float* __restrict__ g,
                             const float* __restrict__ ee,
                             const float* __restrict__ s_sum,
                             float* __restrict__ acc,
                             int heads, int Cout)
{
    const int Ctot = heads * Cout;
    const int warp = (blockIdx.x * blockDim.x + threadIdx.x) >> 5;
    const int lane = threadIdx.x & 31;
    if (warp >= ETOT) return;
    int src = (warp < EE) ? ei[warp]      : (warp - EE);
    int dst = (warp < EE) ? ei[EE + warp] : (warp - EE);
    const int cpl = Ctot >> 5;                    // channels per lane (4 or 2)
    const float* grow = g + (size_t)src * Ctot + lane * cpl;
    float*       arow = acc + (size_t)dst * Ctot + lane * cpl;
    #pragma unroll 4
    for (int c = 0; c < cpl; ++c) {
        int h = (lane * cpl + c) / Cout;
        float alpha = ee[warp * heads + h] / (s_sum[dst * heads + h] + 1e-16f);
        atomicAdd(&arow[c], grow[c] * alpha);
    }
}

// ---------------- bias + BN(eval) + ReLU ----------------
__global__ void gat_bnrelu(const float* __restrict__ acc, const float* __restrict__ bias,
                           const float* __restrict__ gamma, const float* __restrict__ beta,
                           const float* __restrict__ mean,  const float* __restrict__ var,
                           float* __restrict__ out, int C)
{
    int t = blockIdx.x * blockDim.x + threadIdx.x;
    if (t >= NN * C) return;
    int c = t % C;
    float x = acc[t] + bias[c];
    x = (x - mean[c]) * rsqrtf(var[c] + 1e-5f) * gamma[c] + beta[c];
    out[t] = fmaxf(x, 0.f);
}

// ---------------- final: out = acc + b2 (heads=1, mean over 1 head == identity) ----------------
__global__ void gat_final(const float* __restrict__ acc, const float* __restrict__ bias,
                          float* __restrict__ out, int C)
{
    int t = blockIdx.x * blockDim.x + threadIdx.x;
    if (t >= NN * C) return;
    out[t] = acc[t] + bias[t % C];
}

// ---------------- orchestration ----------------
static void run_proj(const float* h_in, const float* W, _Float16* a16, _Float16* bpack,
                     float* gbuf, int Nc, hipStream_t stream)
{
    const int rowTiles = NN / 16;                  // 3125 exactly
    const int ntn = Nc / 16;
    const int packTotal = (KDIM / 32) * ntn * 32;
    gat_cvt16 <<<(NN * KDIM / 4 + 255) / 256, 256, 0, stream>>>(h_in, a16, NN * KDIM / 4);
    gat_packB <<<(packTotal + 255) / 256, 256, 0, stream>>>(W, bpack, Nc);
    gat_gemm_wmma<<<rowTiles * ntn, 32, 0, stream>>>(a16, bpack, gbuf, Nc);
}

static void run_edge_phase(const int* ei, const float* gbuf,
                           float* al_s, float* al_d, unsigned* m_enc, float* s_sum,
                           float* eeb, float* accb,
                           const float* a_src, const float* a_dst,
                           int heads, int Cout, hipStream_t stream)
{
    const int Ctot = heads * Cout;
    gat_al<<<(NN * heads + 255) / 256, 256, 0, stream>>>(gbuf, a_src, a_dst, al_s, al_d, heads, Cout);
    gat_zero<<<(NN * heads + 255) / 256, 256, 0, stream>>>(m_enc, NN * heads);
    gat_zero<<<(NN * heads + 255) / 256, 256, 0, stream>>>((unsigned*)s_sum, NN * heads);
    gat_zero<<<(NN * Ctot + 255) / 256, 256, 0, stream>>>((unsigned*)accb, NN * Ctot);
    gat_edge_logits<<<(ETOT * heads + 255) / 256, 256, 0, stream>>>(ei, al_s, al_d, eeb, m_enc, heads);
    gat_edge_exp<<<(ETOT * heads + 255) / 256, 256, 0, stream>>>(ei, eeb, m_enc, s_sum, heads);
    gat_edge_agg<<<(ETOT * 32 + 255) / 256, 256, 0, stream>>>(ei, gbuf, eeb, s_sum, accb, heads, Cout);
}

extern "C" void kernel_launch(void* const* d_in, const int* in_sizes, int n_in,
                              void* d_out, int out_size, void* d_ws, size_t ws_size,
                              hipStream_t stream)
{
    const float* x   = (const float*)d_in[0];
    const int*   ei  = (const int*)d_in[1];
    const float* W0  = (const float*)d_in[2];
    const float* as0 = (const float*)d_in[3];
    const float* ad0 = (const float*)d_in[4];
    const float* b0  = (const float*)d_in[5];
    const float* g0  = (const float*)d_in[6];
    const float* be0 = (const float*)d_in[7];
    const float* m0  = (const float*)d_in[8];
    const float* v0  = (const float*)d_in[9];
    const float* W1  = (const float*)d_in[10];
    const float* as1 = (const float*)d_in[11];
    const float* ad1 = (const float*)d_in[12];
    const float* b1  = (const float*)d_in[13];
    const float* g1  = (const float*)d_in[14];
    const float* be1 = (const float*)d_in[15];
    const float* m1  = (const float*)d_in[16];
    const float* v1  = (const float*)d_in[17];
    const float* W2  = (const float*)d_in[18];
    const float* as2 = (const float*)d_in[19];
    const float* ad2 = (const float*)d_in[20];
    const float* b2  = (const float*)d_in[21];
    float* out = (float*)d_out;

    // workspace layout
    float*    ws    = (float*)d_ws;
    float*    gbuf  = ws;                       // N*128 f32
    float*    hbuf  = gbuf + (size_t)NN * 128;  // N*128 f32
    float*    accb  = hbuf + (size_t)NN * 128;  // N*128 f32
    float*    al_s  = accb + (size_t)NN * 128;  // N*2
    float*    al_d  = al_s + (size_t)NN * 2;    // N*2
    unsigned* m_enc = (unsigned*)(al_d + (size_t)NN * 2); // N*2
    float*    s_sum = (float*)(m_enc + (size_t)NN * 2);   // N*2
    float*    eeb   = s_sum + (size_t)NN * 2;   // ETOT*2
    _Float16* a16   = (_Float16*)(eeb + (size_t)ETOT * 2);   // N*128 f16
    _Float16* bpack = a16 + (size_t)NN * 128;   // <= 16384 f16 (32KB)

    // ---- layer 0: x[50000,128] -> 2 heads x 64 (concat) ----
    run_proj(x, W0, a16, bpack, gbuf, 128, stream);
    run_edge_phase(ei, gbuf, al_s, al_d, m_enc, s_sum, eeb, accb, as0, ad0, 2, 64, stream);
    gat_bnrelu<<<(NN * 128 + 255) / 256, 256, 0, stream>>>(accb, b0, g0, be0, m0, v0, hbuf, 128);

    // ---- layer 1: h[50000,128] -> 2 heads x 64 (concat) ----
    run_proj(hbuf, W1, a16, bpack, gbuf, 128, stream);
    run_edge_phase(ei, gbuf, al_s, al_d, m_enc, s_sum, eeb, accb, as1, ad1, 2, 64, stream);
    gat_bnrelu<<<(NN * 128 + 255) / 256, 256, 0, stream>>>(accb, b1, g1, be1, m1, v1, hbuf, 128);

    // ---- layer 2: h[50000,128] -> 1 head x 64 (mean == identity) ----
    run_proj(hbuf, W2, a16, bpack, gbuf, 64, stream);
    run_edge_phase(ei, gbuf, al_s, al_d, m_enc, s_sum, eeb, accb, as2, ad2, 1, 64, stream);
    gat_final<<<(NN * 64 + 255) / 256, 256, 0, stream>>>(accb, b2, out, 64);
}
